// PlanEmbeddingNetPredVersion_13700945674651
// MI455X (gfx1250) — matrix operand order repaired and checked by reading, compiled, not run on hardware
//
#include <hip/hip_runtime.h>
#include <math.h>

typedef __attribute__((ext_vector_type(16))) _Float16 v16h;
typedef __attribute__((ext_vector_type(8)))  _Float16 v8h;
typedef __attribute__((ext_vector_type(8)))  float    v8f;

#define B_DIM   128
#define N_DIM   256
#define D_DIM   125
#define OTHER   64
#define MAXP    30
#define EMB     32
#define M_VALID 255      // N-1 conv output positions
#define IDX_PER_B 765    // 3*(N-1)

// ---------------------------------------------------------------------------
// Stage 0: embedding collate.  feature is (B, D, N); output X (B, N, 128) as
// f16 hi/lo pair (compensated split of the fp32 value).
// ---------------------------------------------------------------------------
__global__ void __launch_bounds__(128)
collate_kernel(const float* __restrict__ feature,
               const float* __restrict__ col_embed,
               const float* __restrict__ op_embed,
               _Float16* __restrict__ Xh, _Float16* __restrict__ Xl) {
  int row = blockIdx.x;          // b*N + n
  int b = row >> 8;
  int n = row & 255;
  int c = threadIdx.x;           // 0..127 output channel
  const float* fb = feature + (size_t)b * D_DIM * N_DIM;
  float lengthf = fb[(size_t)(D_DIM - 1) * N_DIM + n];
  int len = (int)lengthf;
  len = len < 0 ? 0 : (len > MAXP ? MAXP : len);
  float val;
  if (c < OTHER) {
    val = fb[(size_t)c * N_DIM + n];
  } else if (c < OTHER + EMB) {
    int e = c - OTHER;
    float s = 0.f;
    for (int p = 0; p < len; ++p) {
      int ci = (int)fb[(size_t)(OTHER + p) * N_DIM + n];
      s += col_embed[ci * EMB + e];
    }
    val = s;
  } else {
    int e = c - (OTHER + EMB);
    float s = 0.f;
    for (int p = 0; p < len; ++p) {
      int oi = (int)fb[(size_t)(OTHER + MAXP + p) * N_DIM + n];
      s += op_embed[oi * EMB + e];
    }
    val = s;
  }
  size_t o = (size_t)row * 128 + c;
  _Float16 h = (_Float16)val;
  Xh[o] = h;
  Xl[o] = (_Float16)(val - (float)h);
}

// ---------------------------------------------------------------------------
// Weight prep: (O, Cin, 3) fp32 -> [k][o][c] f16 hi/lo.
// ---------------------------------------------------------------------------
__global__ void wconv_kernel(const float* __restrict__ W, int O, int Cin,
                             _Float16* __restrict__ Wh, _Float16* __restrict__ Wl) {
  int t = blockIdx.x * blockDim.x + threadIdx.x;
  int total = O * Cin * 3;
  if (t >= total) return;
  int o = t / (Cin * 3);
  int r = t - o * (Cin * 3);
  int c = r / 3;
  int k = r - c * 3;
  float v = W[t];
  _Float16 h = (_Float16)v;
  size_t d = ((size_t)k * O + o) * Cin + c;
  Wh[d] = h;
  Wl[d] = (_Float16)(v - (float)h);
}

// ---------------------------------------------------------------------------
// Tree conv via WMMA.  One wave per 16x16 (o, m) tile of one batch.
// out[o, m] = sum_{k=0..2} sum_c W[o,c,k] * X[idx[3m+k], c]   (+bias)
// Compensated f16: acc += Ah*Bh + Ah*Bl + Al*Bh  (f32 accumulate).
// Result stored node-major: Y[b, m+1, o]; node 0 zeroed separately.
// ---------------------------------------------------------------------------
__global__ void __launch_bounds__(32)
tree_conv_wmma_kernel(const _Float16* __restrict__ Xh, const _Float16* __restrict__ Xl,
                      const _Float16* __restrict__ Wh, const _Float16* __restrict__ Wl,
                      const float* __restrict__ bias,
                      const long long* __restrict__ idx,
                      float* __restrict__ Y, int Cin, int Cout) {
  const int lane  = threadIdx.x;   // 0..31 (wave32)
  const int b     = blockIdx.z;
  const int o0    = blockIdx.y * 16;
  const int m0    = blockIdx.x * 16;
  const int nloc  = lane & 15;     // D column / A row within tile
  const int khalf = lane >> 4;

  v8f acc = {};
  const long long* idxb = idx + (size_t)b * IDX_PER_B;
  const int m  = m0 + nloc;
  const int mi = m < M_VALID ? m : (M_VALID - 1);   // clamp (keeps EXEC uniform)

  for (int k = 0; k < 3; ++k) {
    int node = (int)idxb[3 * mi + k];
    const _Float16* xhrow = Xh + ((size_t)b * N_DIM + node) * Cin;
    const _Float16* xlrow = Xl + ((size_t)b * N_DIM + node) * Cin;
    const _Float16* whrow = Wh + ((size_t)k * Cout + (o0 + nloc)) * Cin;
    const _Float16* wlrow = Wl + ((size_t)k * Cout + (o0 + nloc)) * Cin;
    for (int c0 = 0; c0 < Cin; c0 += 32) {
      // A fragment: lane holds row nloc; K map = {khalf*8+h} and {16+khalf*8+h}
      union { v16h v; v8h h2[2]; } ah, al, bh, bl;
      // B fragment: lane holds col nloc; K map = {khalf*8+h} and {16+khalf*8+h}
      bh.h2[0] = *(const v8h*)(xhrow + c0 + khalf * 8);
      bh.h2[1] = *(const v8h*)(xhrow + c0 + 16 + khalf * 8);
      bl.h2[0] = *(const v8h*)(xlrow + c0 + khalf * 8);
      bl.h2[1] = *(const v8h*)(xlrow + c0 + 16 + khalf * 8);
      ah.h2[0] = *(const v8h*)(whrow + c0 + khalf * 8);
      ah.h2[1] = *(const v8h*)(whrow + c0 + 16 + khalf * 8);
      al.h2[0] = *(const v8h*)(wlrow + c0 + khalf * 8);
      al.h2[1] = *(const v8h*)(wlrow + c0 + 16 + khalf * 8);
      acc = __builtin_amdgcn_wmma_f32_16x16x32_f16(false, ah.v, false, bh.v, (short)0, acc, false, false);
      acc = __builtin_amdgcn_wmma_f32_16x16x32_f16(false, ah.v, false, bl.v, (short)0, acc, false, false);
      acc = __builtin_amdgcn_wmma_f32_16x16x32_f16(false, al.v, false, bh.v, (short)0, acc, false, false);
    }
  }

  if (m < M_VALID) {
    int node = m + 1;                              // zero column prepended at node 0
    float* yrow = Y + ((size_t)b * N_DIM + node) * Cout + o0 + khalf * 8;
    const float* brow = bias + o0 + khalf * 8;
#pragma unroll
    for (int v = 0; v < 8; ++v) yrow[v] = acc[v] + brow[v];  // rows o = o0+8*khalf+v
  }
}

__global__ void zerocol_kernel(float* __restrict__ Y, int Cout) {
  int b = blockIdx.x;
  for (int c = threadIdx.x; c < Cout; c += blockDim.x)
    Y[(size_t)b * N_DIM * Cout + c] = 0.f;
}

// ---------------------------------------------------------------------------
// Per-batch global norm (mean/std over all C*N, ddof=1) + optional leaky,
// in-place on fp32 Y; optionally emits f16 hi/lo copies for next layer.
// ---------------------------------------------------------------------------
__global__ void __launch_bounds__(256)
tree_norm_kernel(float* __restrict__ Y, int Cout, int do_leaky,
                 _Float16* __restrict__ Xh, _Float16* __restrict__ Xl) {
  __shared__ float ssum[256];
  __shared__ float ssq[256];
  int b = blockIdx.x;
  int tid = threadIdx.x;
  size_t cnt = (size_t)N_DIM * Cout;
  float* yb = Y + (size_t)b * cnt;
  float s = 0.f, sq = 0.f;
  for (size_t i = tid; i < cnt; i += 256) { float v = yb[i]; s += v; sq += v * v; }
  ssum[tid] = s; ssq[tid] = sq;
  __syncthreads();
  for (int st = 128; st > 0; st >>= 1) {
    if (tid < st) { ssum[tid] += ssum[tid + st]; ssq[tid] += ssq[tid + st]; }
    __syncthreads();
  }
  float mean = ssum[0] / (float)cnt;
  float var  = (ssq[0] - ssum[0] * mean) / (float)(cnt - 1);
  var = var > 0.f ? var : 0.f;
  float scale = 1.f / (sqrtf(var) + 1e-5f);
  for (size_t i = tid; i < cnt; i += 256) {
    float v = (yb[i] - mean) * scale;
    if (do_leaky) v = v >= 0.f ? v : 0.01f * v;
    yb[i] = v;
    if (Xh) {
      _Float16 h = (_Float16)v;
      size_t o = (size_t)b * cnt + i;
      Xh[o] = h;
      Xl[o] = (_Float16)(v - (float)h);
    }
  }
}

// ---------------------------------------------------------------------------
// Max-pool over nodes (C=64) + 64->32 linear.
// ---------------------------------------------------------------------------
__global__ void __launch_bounds__(64)
final_kernel(const float* __restrict__ Y, const float* __restrict__ Wlin,
             const float* __restrict__ blin, float* __restrict__ out) {
  __shared__ float pooled[64];
  int b = blockIdx.x;
  int tid = threadIdx.x;
  {
    float mx = -3.402823466e38f;
    const float* yb = Y + (size_t)b * N_DIM * 64 + tid;
    for (int n = 0; n < N_DIM; ++n) mx = fmaxf(mx, yb[(size_t)n * 64]);
    pooled[tid] = mx;
  }
  __syncthreads();
  if (tid < 32) {
    float s = blin[tid];
    for (int c = 0; c < 64; ++c) s += pooled[c] * Wlin[tid * 64 + c];
    out[b * 32 + tid] = s;
  }
}

// ---------------------------------------------------------------------------
extern "C" void kernel_launch(void* const* d_in, const int* in_sizes, int n_in,
                              void* d_out, int out_size, void* d_ws, size_t ws_size,
                              hipStream_t stream) {
  (void)in_sizes; (void)n_in; (void)out_size; (void)ws_size;
  const float*     feature = (const float*)d_in[0];
  const long long* idx     = (const long long*)d_in[1];
  const float*     col_emb = (const float*)d_in[2];
  const float*     op_emb  = (const float*)d_in[3];
  const float*     W1 = (const float*)d_in[4];
  const float*     b1 = (const float*)d_in[5];
  const float*     W2 = (const float*)d_in[6];
  const float*     b2 = (const float*)d_in[7];
  const float*     W3 = (const float*)d_in[8];
  const float*     b3 = (const float*)d_in[9];
  const float*     Wlin = (const float*)d_in[10];
  const float*     blin = (const float*)d_in[11];
  float* out = (float*)d_out;

  const size_t nW1 = 256 * 128 * 3, nW2 = 128 * 256 * 3, nW3 = 64 * 128 * 3;
  const size_t nWall = nW1 + nW2 + nW3;                 // 221184 halves
  char* ws = (char*)d_ws;
  size_t off = 0;
  _Float16* Wh_ws = (_Float16*)(ws + off); off += (nWall * 2 + 255) & ~(size_t)255;
  _Float16* Wl_ws = (_Float16*)(ws + off); off += (nWall * 2 + 255) & ~(size_t)255;
  const size_t nXmax = (size_t)B_DIM * N_DIM * 256;     // max C = 256
  _Float16* Xh = (_Float16*)(ws + off); off += nXmax * 2;
  _Float16* Xl = (_Float16*)(ws + off); off += nXmax * 2;
  float*    Yb = (float*)(ws + off);    off += nXmax * 4;

  // weight prep + collate
  wconv_kernel<<<(int)((nW1 + 255) / 256), 256, 0, stream>>>(W1, 256, 128, Wh_ws, Wl_ws);
  wconv_kernel<<<(int)((nW2 + 255) / 256), 256, 0, stream>>>(W2, 128, 256, Wh_ws + nW1, Wl_ws + nW1);
  wconv_kernel<<<(int)((nW3 + 255) / 256), 256, 0, stream>>>(W3, 64, 128, Wh_ws + nW1 + nW2, Wl_ws + nW1 + nW2);
  collate_kernel<<<B_DIM * N_DIM, 128, 0, stream>>>(feature, col_emb, op_emb, Xh, Xl);

  // layer 1: Cin=128 -> Cout=256
  tree_conv_wmma_kernel<<<dim3(16, 16, B_DIM), 32, 0, stream>>>(Xh, Xl, Wh_ws, Wl_ws, b1, idx, Yb, 128, 256);
  zerocol_kernel<<<B_DIM, 256, 0, stream>>>(Yb, 256);
  tree_norm_kernel<<<B_DIM, 256, 0, stream>>>(Yb, 256, 1, Xh, Xl);

  // layer 2: Cin=256 -> Cout=128
  tree_conv_wmma_kernel<<<dim3(16, 8, B_DIM), 32, 0, stream>>>(Xh, Xl, Wh_ws + nW1, Wl_ws + nW1, b2, idx, Yb, 256, 128);
  zerocol_kernel<<<B_DIM, 128, 0, stream>>>(Yb, 128);
  tree_norm_kernel<<<B_DIM, 256, 0, stream>>>(Yb, 128, 1, Xh, Xl);

  // layer 3: Cin=128 -> Cout=64 (norm only, no leaky, no f16 emit)
  tree_conv_wmma_kernel<<<dim3(16, 4, B_DIM), 32, 0, stream>>>(Xh, Xl, Wh_ws + nW1 + nW2, Wl_ws + nW1 + nW2, b3, idx, Yb, 128, 64);
  zerocol_kernel<<<B_DIM, 64, 0, stream>>>(Yb, 64);
  tree_norm_kernel<<<B_DIM, 256, 0, stream>>>(Yb, 64, 0, nullptr, nullptr);

  // pool + linear head
  final_kernel<<<B_DIM, 64, 0, stream>>>(Yb, Wlin, blin, out);
}
